// CRFDecoder_61813169324327
// MI455X (gfx1250) — compile-verified, matrix operand order checked by reading, not run on hardware
//
#include <hip/hip_runtime.h>
#include <math.h>

#define SEQ_LEN 256
#define BATCH   32
#define HIDDEN  512
#define VOCAB   1024
#define NBLK    16          // blocks in persistent CRF kernel

typedef __attribute__((ext_vector_type(16))) __bf16          v16bf;
typedef __attribute__((ext_vector_type(8)))  float           v8f;
typedef __attribute__((ext_vector_type(8)))  unsigned short  us8;

union FragAB { v16bf v; us8 h[2]; unsigned u[8]; unsigned short s[16]; };
union FragC  { v8f   v; float f[8]; };

// f32 -> bf16 via native conversion (backend picks v_cvt_* when available)
__device__ __forceinline__ unsigned short f2bf(float x) {
  union { __bf16 b; unsigned short u; } c; c.b = (__bf16)x; return c.u;
}
__device__ __forceinline__ unsigned pack_bf16(float lo, float hi) {
  union { __bf16 b[2]; unsigned u; } c;
  c.b[0] = (__bf16)lo; c.b[1] = (__bf16)hi;
  return c.u;
}

// per-lane 16B global -> LDS async DMA (ASYNCcnt-tracked)
__device__ __forceinline__ void async_g2l_b128(const void* gaddr, const void* lds) {
  unsigned l = (unsigned)(uintptr_t)lds;                 // LDS byte offset = low 32 bits
  unsigned long long g = (unsigned long long)(uintptr_t)gaddr;
  asm volatile("global_load_async_to_lds_b128 %0, %1, off" :: "v"(l), "v"(g) : "memory");
}
__device__ __forceinline__ void wait_async0() {
  asm volatile("s_wait_asynccnt 0" ::: "memory");
}

// ---------------------------------------------------------------------------
// Kernel 1: logits = enc(8192x512) @ W^T(512x1024) + bias, bf16 WMMA f32-acc.
// Double-buffered async-DMA pipeline: global_load_async_to_lds_b128 fills the
// next K-tile while the current tile feeds the WMMA.
// ---------------------------------------------------------------------------
__global__ __launch_bounds__(256) void gemm_logits(
    const float* __restrict__ enc, const float* __restrict__ W,
    const float* __restrict__ bias, float* __restrict__ logits)
{
  // 36-float row stride (144B) keeps all 8-float chunks 16B-aligned
  __shared__ __align__(16) float Af[2][128][36];
  __shared__ __align__(16) float Bf[2][16][36];

  const int tid  = threadIdx.x;
  const int lane = tid & 31, wave = tid >> 5;
  const int m0   = blockIdx.x * 128;
  const int n0   = blockIdx.y * 16;
  const int ln   = lane & 15, hi = lane >> 4;

  auto issue = [&](int k0, int buf) {
    #pragma unroll
    for (int i = 0; i < 4; ++i) {
      int idx = (tid + 256 * i) * 4;         // float index in 128x32 tile
      int r = idx >> 5, c = idx & 31;
      async_g2l_b128(enc + (size_t)(m0 + r) * HIDDEN + k0 + c, &Af[buf][r][c]);
    }
    if (tid < 128) {                          // waves 0-3
      int idx = tid * 4;
      int n = idx >> 5, c = idx & 31;
      async_g2l_b128(W + (size_t)(n0 + n) * HIDDEN + k0 + c, &Bf[buf][n][c]);
    }
  };

  FragC acc = {};
  issue(0, 0);
  int cur = 0;
  for (int k0 = 0; k0 < HIDDEN; k0 += 32, cur ^= 1) {
    wait_async0();          // my wave's DMA into buf[cur] complete
    __syncthreads();        // all waves' DMA complete; prior reads of buf[cur^1] done
    if (k0 + 32 < HIDDEN) issue(k0 + 32, cur ^ 1);

    FragAB a, b;
    const float* ar = &Af[cur][16 * wave + ln][0];
    const float* br = &Bf[cur][ln][0];
    #pragma unroll
    for (int j = 0; j < 4; ++j) {
      a.u[j]     = pack_bf16(ar[hi * 8 + 2 * j],      ar[hi * 8 + 2 * j + 1]);
      a.u[4 + j] = pack_bf16(ar[16 + hi * 8 + 2 * j], ar[16 + hi * 8 + 2 * j + 1]);
      b.u[j]     = pack_bf16(br[hi * 8 + 2 * j],      br[hi * 8 + 2 * j + 1]);
      b.u[4 + j] = pack_bf16(br[16 + hi * 8 + 2 * j], br[16 + hi * 8 + 2 * j + 1]);
    }
    acc.v = __builtin_amdgcn_wmma_f32_16x16x32_bf16(false, a.v, false, b.v,
                                                    (short)0, acc.v, false, false);
  }

  #pragma unroll
  for (int r = 0; r < 8; ++r) {
    int M = m0 + 16 * wave + r + hi * 8;
    int N = n0 + ln;
    logits[(size_t)M * VOCAB + N] = acc.f[r] + bias[N];
  }
}

// ---------------------------------------------------------------------------
// Kernel 2: Et[v][u] = bf16(exp(transition[u][v]))  (transposed, L2-resident)
// ---------------------------------------------------------------------------
__global__ __launch_bounds__(256) void prep_Et(const float* __restrict__ tr,
                                               unsigned short* __restrict__ Et)
{
  int u = blockIdx.x;
  for (int v = threadIdx.x; v < VOCAB; v += 256)
    Et[(size_t)v * VOCAB + u] = f2bf(__expf(tr[(size_t)u * VOCAB + v]));
}

// Kernel 3: Z0 = logits[0]; zero the device barrier counter
__global__ void init_Z(const float* __restrict__ logits, float* __restrict__ Z,
                       int* __restrict__ bar)
{
  int i = blockIdx.x * blockDim.x + threadIdx.x;
  if (i < BATCH * VOCAB) Z[i] = logits[i];
  if (i == 0) *bar = 0;
}

// device-wide barrier (counter-based, release/acquire, WGP$ fenced)
__device__ __forceinline__ void gbar(int* bar, int expected) {
  __syncthreads();
  if (threadIdx.x == 0) {
    __threadfence();
    __hip_atomic_fetch_add(bar, 1, __ATOMIC_RELEASE, __HIP_MEMORY_SCOPE_AGENT);
    while (__hip_atomic_load(bar, __ATOMIC_ACQUIRE, __HIP_MEMORY_SCOPE_AGENT) < expected)
      __builtin_amdgcn_s_sleep(1);
    __threadfence();
  }
  __syncthreads();
}

// ---------------------------------------------------------------------------
// Kernel 4 (persistent, 16 blocks x 8 waves): all 255 CRF steps + finalize.
// Et rows for this block's 8 tiles (256KB) are DMA'd into LDS ONCE, then all
// B fragments are served from LDS for the whole 255-step recursion.
// ---------------------------------------------------------------------------
#define ET_LDS_BYTES (128 * 1024 * 2)        // 128 rows x 1024 bf16
#define CRF_LDS_BYTES (ET_LDS_BYTES + 256 * 4)

__global__ __launch_bounds__(256) void crf_persist(
    unsigned short* __restrict__ P, const unsigned short* __restrict__ Et,
    float* __restrict__ mbuf, const float* __restrict__ logits,
    const int* __restrict__ lengths, float* __restrict__ Z0,
    float* __restrict__ Z1, int* __restrict__ bar, float* __restrict__ perb,
    float* __restrict__ out, const int* __restrict__ tgt,
    const float* __restrict__ tr)
{
  extern __shared__ __align__(16) char smem[];
  unsigned short* EtL = (unsigned short*)smem;           // [128][1024] bf16
  float* red = (float*)(smem + ET_LDS_BYTES);            // [256]

  const int blk  = blockIdx.x;            // 0..15
  const int tid  = threadIdx.x;
  const int lane = tid & 31, wave = tid >> 5;
  const int ln   = lane & 15, hi = lane >> 4;

  const int tile = blk * 8 + wave;        // 0..127
  const int mt = tile >> 6, nt = tile & 63;

  // ---- one-time: DMA this block's 128 Et rows (8 tiles x 16 rows) into LDS
  for (int i = tid; i < 16384; i += 256) {          // 16384 b128 chunks
    int row = i >> 7;                               // 0..127 (row 16w+j = wave w)
    int c8  = (i & 127) * 8;                        // ushort offset in row
    int gn  = ((blk & 7) * 8 + (row >> 4)) * 16 + (row & 15);
    async_g2l_b128(Et + (size_t)gn * VOCAB + c8, EtL + row * 1024 + c8);
  }
  wait_async0();
  __syncthreads();

  const unsigned short* eL = EtL + (wave * 16 + ln) * 1024;   // my B rows (LDS)
  const unsigned short* prow = P + (size_t)(mt * 16 + ln) * VOCAB;

  int mylen[8];
  #pragma unroll
  for (int r = 0; r < 8; ++r) mylen[r] = lengths[mt * 16 + r + hi * 8];

  int epoch = 0;
  float* zp = Z0;
  float* zn = Z1;

  for (int t = 1; t < SEQ_LEN; ++t) {
    // ---- phase 1: rows 2*blk, 2*blk+1: rowmax + exp (packed bf16 stores)
    for (int rr = 0; rr < 2; ++rr) {
      int b = 2 * blk + rr;
      const float* zr = zp + b * VOCAB;
      float mx = -3.4e38f;
      #pragma unroll
      for (int i = 0; i < 4; ++i) mx = fmaxf(mx, zr[tid + 256 * i]);
      red[tid] = mx; __syncthreads();
      for (int s = 128; s > 0; s >>= 1) {
        if (tid < s) red[tid] = fmaxf(red[tid], red[tid + s]);
        __syncthreads();
      }
      float m = red[0]; __syncthreads();
      unsigned* Pp = (unsigned*)(P + b * VOCAB);
      #pragma unroll
      for (int i = 0; i < 2; ++i) {
        int v2 = tid + 256 * i;                    // u32 index 0..511
        float z0 = zr[2 * v2], z1 = zr[2 * v2 + 1];
        Pp[v2] = pack_bf16(__expf(z0 - m), __expf(z1 - m));
      }
      if (tid == 0) mbuf[b] = m;
      __syncthreads();
    }
    gbar(bar, NBLK * (++epoch));

    // ---- phase 2: one 16x16 WMMA tile per wave, B from LDS, dual accums
    {
      const float* lg = logits + (size_t)t * BATCH * VOCAB;
      #pragma unroll
      for (int r = 0; r < 8; ++r)      // warm the epilogue lines while we matmul
        __builtin_prefetch(lg + (mt * 16 + r + hi * 8) * VOCAB + nt * 16, 0, 0);

      FragC acc0 = {}, acc1 = {};
      #pragma unroll 4
      for (int k0 = 0; k0 < VOCAB; k0 += 64) {
        FragAB a0, b0, a1, b1;
        a0.h[0] = *(const us8*)(prow + k0 + hi * 8);
        a0.h[1] = *(const us8*)(prow + k0 + 16 + hi * 8);
        b0.h[0] = *(const us8*)(eL + k0 + hi * 8);
        b0.h[1] = *(const us8*)(eL + k0 + 16 + hi * 8);
        acc0.v = __builtin_amdgcn_wmma_f32_16x16x32_bf16(false, a0.v, false, b0.v,
                                                         (short)0, acc0.v, false, false);
        a1.h[0] = *(const us8*)(prow + k0 + 32 + hi * 8);
        a1.h[1] = *(const us8*)(prow + k0 + 48 + hi * 8);
        b1.h[0] = *(const us8*)(eL + k0 + 32 + hi * 8);
        b1.h[1] = *(const us8*)(eL + k0 + 48 + hi * 8);
        acc1.v = __builtin_amdgcn_wmma_f32_16x16x32_bf16(false, a1.v, false, b1.v,
                                                         (short)0, acc1.v, false, false);
      }
      #pragma unroll
      for (int r = 0; r < 8; ++r) {
        int brow = mt * 16 + r + hi * 8;   // batch index 0..31
        int v = nt * 16 + ln;
        float s = acc0.f[r] + acc1.f[r];
        float z = (t < mylen[r])
                    ? (mbuf[brow] + __logf(s) + lg[brow * VOCAB + v])
                    : zp[brow * VOCAB + v];
        zn[brow * VOCAB + v] = z;
      }
    }
    gbar(bar, NBLK * (++epoch));
    float* tmp = zp; zp = zn; zn = tmp;
  }

  // ---- finalize: rows 2*blk, 2*blk+1: logsumexp(Zfinal) - gold score
  for (int rr = 0; rr < 2; ++rr) {
    int b = 2 * blk + rr;
    int lenb = lengths[b];
    float mx = -3.4e38f;
    for (int v = tid; v < VOCAB; v += 256) mx = fmaxf(mx, zp[b * VOCAB + v]);
    red[tid] = mx; __syncthreads();
    for (int s = 128; s > 0; s >>= 1) {
      if (tid < s) red[tid] = fmaxf(red[tid], red[tid + s]);
      __syncthreads();
    }
    float m = red[0]; __syncthreads();
    float se = 0.f;
    for (int v = tid; v < VOCAB; v += 256) se += __expf(zp[b * VOCAB + v] - m);
    red[tid] = se; __syncthreads();
    for (int s = 128; s > 0; s >>= 1) {
      if (tid < s) red[tid] += red[tid + s];
      __syncthreads();
    }
    float logZ = m + __logf(red[0]); __syncthreads();

    float sc = 0.f;
    for (int tt = tid; tt < SEQ_LEN; tt += 256) {
      if (tt < lenb) {
        int y = tgt[tt * BATCH + b];
        sc += logits[((size_t)tt * BATCH + b) * VOCAB + y];
        if (tt >= 1) {
          int yp = tgt[(tt - 1) * BATCH + b];
          sc += tr[(size_t)yp * VOCAB + y];
        }
      }
    }
    red[tid] = sc; __syncthreads();
    for (int s = 128; s > 0; s >>= 1) {
      if (tid < s) red[tid] += red[tid + s];
      __syncthreads();
    }
    if (tid == 0) perb[b] = logZ - red[0];
    __syncthreads();
  }
  gbar(bar, NBLK * (++epoch));
  if (blk == 0 && tid == 0) {
    float s = 0.f;
    for (int i = 0; i < BATCH; ++i) s += perb[i];
    out[0] = s / (float)BATCH;
  }
}

// ---------------------------------------------------------------------------
extern "C" void kernel_launch(void* const* d_in, const int* in_sizes, int n_in,
                              void* d_out, int out_size, void* d_ws, size_t ws_size,
                              hipStream_t stream)
{
  const float* enc  = (const float*)d_in[0];   // (256,32,512)
  const float* W    = (const float*)d_in[1];   // (1024,512)
  const float* bias = (const float*)d_in[2];   // (1024,)
  const float* tr   = (const float*)d_in[3];   // (1024,1024)
  const int*   tgt  = (const int*)d_in[4];     // (256,32)
  const int*   len  = (const int*)d_in[5];     // (32,)

  char* ws = (char*)d_ws;
  float*          logits = (float*)ws;                          // 33554432 B
  unsigned short* Et     = (unsigned short*)(ws + 33554432);    //  2097152 B
  float*          Z0     = (float*)(ws + 35651584);             //   131072 B
  float*          Z1     = (float*)(ws + 35782656);             //   131072 B
  unsigned short* P      = (unsigned short*)(ws + 35913728);    //    65536 B
  float*          mbuf   = (float*)(ws + 35979264);             //      128 B
  float*          perb   = (float*)(ws + 35979392);             //      128 B
  int*            bar    = (int*)(ws + 35979520);               //      128 B

  gemm_logits<<<dim3(64, 64), 256, 0, stream>>>(enc, W, bias, logits);
  prep_Et<<<VOCAB, 256, 0, stream>>>(tr, Et);
  init_Z<<<(BATCH * VOCAB + 255) / 256, 256, 0, stream>>>(logits, Z0, bar);

  crf_persist<<<NBLK, 256, CRF_LDS_BYTES, stream>>>(P, Et, mbuf, logits, len,
                                                    Z0, Z1, bar, perb,
                                                    (float*)d_out, tgt, tr);
}
